// Self_Attention_75694503625032
// MI455X (gfx1250) — matrix operand order repaired and checked by reading, not compiled, not run on hardware
//
#include <hip/hip_runtime.h>

#define B_  4
#define C_  256
#define C8_ 32
#define N_  4096
#define HH_STRIDE 40   // halves per LDS row of the hh tile (32 data + 8 pad)

typedef __attribute__((ext_vector_type(16))) _Float16 v16h;
typedef __attribute__((ext_vector_type(8)))  _Float16 v8h;
typedef __attribute__((ext_vector_type(8)))  float    v8f;
typedef __attribute__((ext_vector_type(4)))  int      v4i;

// typed, address-space-qualified pointers for the async-copy builtin
typedef __attribute__((address_space(1))) v4i g_v4i;   // global int4
typedef __attribute__((address_space(3))) v4i l_v4i;   // LDS int4

union AFrag { v16h v; v8h h8[2]; };

static __device__ __forceinline__ v8f wmma_f16(v16h a, v16h b, v8f c) {
  // D = A(16x32 f16) * B(32x16 f16) + C(16x16 f32)
  return __builtin_amdgcn_wmma_f32_16x16x32_f16(false, a, false, b, (short)0, c,
                                                false, false);
}

// ---------------------------------------------------------------------------
// Kernel 1a: x (B,C,N) f32 -> xhT (B,N,C) f16   (transposed for B-fragments)
// ---------------------------------------------------------------------------
__global__ void cvt_x_kernel(const float* __restrict__ x, _Float16* __restrict__ xhT) {
  size_t idx = (size_t)blockIdx.x * blockDim.x + threadIdx.x;
  if (idx >= (size_t)B_ * C_ * N_) return;
  int n = (int)(idx % N_);
  int c = (int)((idx / N_) % C_);
  int b = (int)(idx / ((size_t)C_ * N_));
  xhT[((size_t)b * N_ + n) * C_ + c] = (_Float16)x[idx];
}

// ---------------------------------------------------------------------------
// Kernel 1b: pack [wf; wg; wh] -> W16 (320x256 f16), biases -> b320 (f32)
// ---------------------------------------------------------------------------
__global__ void cvt_w_kernel(const float* __restrict__ wf, const float* __restrict__ bf,
                             const float* __restrict__ wg, const float* __restrict__ bg,
                             const float* __restrict__ wh, const float* __restrict__ bh,
                             _Float16* __restrict__ W16, float* __restrict__ b320) {
  int idx = blockIdx.x * blockDim.x + threadIdx.x;
  if (idx < 320 * 256) {
    int o = idx >> 8, c = idx & 255;
    float v = (o < 32) ? wf[o * 256 + c]
            : (o < 64) ? wg[(o - 32) * 256 + c]
                       : wh[(o - 64) * 256 + c];
    W16[idx] = (_Float16)v;
  }
  if (idx < 320)
    b320[idx] = (idx < 32) ? bf[idx] : (idx < 64) ? bg[idx - 32] : bh[idx - 64];
}

// ---------------------------------------------------------------------------
// Kernel 2: projections, double-buffered A-fragment pipeline.
//   Per block: batch b, 64 pixels; 4 waves x 16 pixels; o loops over 320 rows.
// ---------------------------------------------------------------------------
__global__ __launch_bounds__(128) void proj_kernel(
    const _Float16* __restrict__ xhT, const _Float16* __restrict__ W16,
    const float* __restrict__ b320,
    _Float16* __restrict__ fT, _Float16* __restrict__ gT,
    _Float16* __restrict__ hh16) {
  const int lane = threadIdx.x & 31;
  const int wv   = threadIdx.x >> 5;
  const int col  = lane & 15;   // lane % 16
  const int hv   = lane >> 4;   // lane half
  const int b    = blockIdx.y;
  const int n0   = blockIdx.x * 64 + wv * 16;

  // Preload B fragments: x^T tile, K = c (8 chunks of 32) — stays in regs.
  v16h xb[8];
  const _Float16* xrow = xhT + ((size_t)b * N_ + n0 + col) * C_;
#pragma unroll
  for (int k = 0; k < 8; ++k)
    xb[k] = *reinterpret_cast<const v16h*>(xrow + k * 32 + 16 * hv);

  auto load_a = [&](AFrag* dst, int o0) {
    const _Float16* wrow = W16 + (size_t)(o0 + col) * C_;
#pragma unroll
    for (int k = 0; k < 8; ++k) {
      dst[k].h8[0] = *reinterpret_cast<const v8h*>(wrow + k * 32 + 16 * hv);
      dst[k].h8[1] = *reinterpret_cast<const v8h*>(wrow + k * 32 + 16 * hv + 8);
    }
  };
  auto compute_store = [&](const AFrag* a, int o0) {
    v8f acc = {};
#pragma unroll
    for (int k = 0; k < 8; ++k) acc = wmma_f16(a[k].v, xb[k], acc);
#pragma unroll
    for (int j = 0; j < 8; ++j) {
      int oo = o0 + 8 * hv + j;          // D element: M = 8*hv + j
      int nn = n0 + col;                 // N = lane % 16
      float v = acc[j] + b320[oo];
      if (oo < 32)
        fT[((size_t)b * N_ + nn) * C8_ + oo] = (_Float16)v;
      else if (oo < 64)
        gT[((size_t)b * N_ + nn) * C8_ + (oo - 32)] = (_Float16)v;
      else
        hh16[((size_t)b * C_ + (oo - 64)) * N_ + nn] = (_Float16)v;
    }
  };

  AFrag a0[8], a1[8];
  load_a(a0, 0);
  for (int o0 = 0; o0 < 320; o0 += 32) {
    if (o0 + 16 < 320) load_a(a1, o0 + 16);   // prefetch while computing a0
    compute_store(a0, o0);
    if (o0 + 32 < 320) load_a(a0, o0 + 32);   // prefetch while computing a1
    compute_store(a1, o0 + 16);
  }
}

// ---------------------------------------------------------------------------
// Kernel 3: fused flash attention.
//   Block = (batch b, 64 query rows), 4 waves x 16 rows, lockstep over m.
//   hh tile (256 x 32 f16, 16 KB) staged block-cooperatively into LDS with
//   CDNA5 async global->LDS copies, double-buffered against compute.
// ---------------------------------------------------------------------------
__global__ __launch_bounds__(128) void attn_kernel(
    const _Float16* __restrict__ fT, const _Float16* __restrict__ gT,
    const _Float16* __restrict__ hh16, const float* __restrict__ x,
    const float* __restrict__ gamma, float* __restrict__ out) {
  __shared__ _Float16 hh_lds[2][256 * HH_STRIDE];  // 2 x 20 KB
  __shared__ _Float16 lds_p[4][16 * 32];           // 4 KB  (P per wave)
  const int tid  = threadIdx.x;
  const int lane = tid & 31;
  const int wv   = tid >> 5;
  const int col  = lane & 15;
  const int hv   = lane >> 4;
  const int b    = blockIdx.y;
  const int n0   = blockIdx.x * 64 + wv * 16;

  const _Float16* hbase = hh16 + (size_t)b * C_ * N_;
  const _Float16* gbase = gT + (size_t)b * N_ * C8_;

  // --- block-cooperative hh tile copy: 1024 chunks of 16B, 8 per thread ---
  auto load_hh = [&](int m0, int buf) {
#pragma unroll
    for (int r = 0; r < 8; ++r) {
      int id = r * 128 + tid;              // 0..1023
      int c  = id >> 2;                    // channel row 0..255
      int mp = id & 3;                     // 16B sub-chunk of the 32-m row
      const _Float16* gp = hbase + (size_t)c * N_ + m0 + mp * 8;
      _Float16*       lp = &hh_lds[buf][c * HH_STRIDE + mp * 8];
#if __has_builtin(__builtin_amdgcn_global_load_async_to_lds_b128)
      __builtin_amdgcn_global_load_async_to_lds_b128(
          (g_v4i*)gp, (l_v4i*)lp, 0, 0);
#else
      *reinterpret_cast<v8h*>(lp) = *reinterpret_cast<const v8h*>(gp);
#endif
    }
  };
  auto wait_async = [&]() {
#if __has_builtin(__builtin_amdgcn_global_load_async_to_lds_b128)
#if __has_builtin(__builtin_amdgcn_s_wait_asynccnt)
    __builtin_amdgcn_s_wait_asynccnt(0);
#else
    asm volatile("s_wait_asynccnt 0" ::: "memory");
#endif
#endif
  };

  // kick off first hh tile; its latency hides under all of pass 1
  load_hh(0, 0);

  // A fragment of f^T (rows = query n, K = o in [0,32)) — lives in regs.
  AFrag af;
  const _Float16* frow = fT + ((size_t)b * N_ + n0 + col) * C8_;
  af.h8[0] = *reinterpret_cast<const v8h*>(frow + 16 * hv);
  af.h8[1] = *reinterpret_cast<const v8h*>(frow + 16 * hv + 8);

  // ---- pass 1: row max over all m (two independent WMMA chains/iter) ----
  float rm[8];
#pragma unroll
  for (int j = 0; j < 8; ++j) rm[j] = -3.0e38f;
  for (int m0 = 0; m0 < N_; m0 += 32) {
    v16h bg0 = *reinterpret_cast<const v16h*>(gbase + (size_t)(m0 + col) * C8_ + 16 * hv);
    v16h bg1 = *reinterpret_cast<const v16h*>(gbase + (size_t)(m0 + 16 + col) * C8_ + 16 * hv);
    v8f z = {};
    v8f s0 = wmma_f16(af.v, bg0, z);
    v8f s1 = wmma_f16(af.v, bg1, z);
#pragma unroll
    for (int j = 0; j < 8; ++j) rm[j] = fmaxf(rm[j], fmaxf(s0[j], s1[j]));
  }
#pragma unroll
  for (int j = 0; j < 8; ++j) {
#pragma unroll
    for (int off = 1; off < 16; off <<= 1)
      rm[j] = fmaxf(rm[j], __shfl_xor(rm[j], off, 16));
  }

  // ---- pass 2: accumulate O[16 rows x 256 chan], unnormalized ----
  v8f zero = {};
  v8f oacc[16];
#pragma unroll
  for (int cb = 0; cb < 16; ++cb) oacc[cb] = zero;
  float rs[8];
#pragma unroll
  for (int j = 0; j < 8; ++j) rs[j] = 0.f;

  _Float16* pbuf = &lds_p[wv][0];

  wait_async();
  __syncthreads();        // first hh tile resident for all waves

  int cur = 0;
  for (int m0 = 0; m0 < N_; m0 += 32) {
    if (m0 + 32 < N_) load_hh(m0 + 32, cur ^ 1);   // async prefetch next tile

    // S tile (16x32) -> P = exp(S - rowmax), staged to per-wave LDS buffer
#pragma unroll
    for (int t = 0; t < 2; ++t) {
      v16h bg = *reinterpret_cast<const v16h*>(
          gbase + (size_t)(m0 + 16 * t + col) * C8_ + 16 * hv);
      v8f z = {};
      v8f s = wmma_f16(af.v, bg, z);
#pragma unroll
      for (int j = 0; j < 8; ++j) {
        float e = __expf(s[j] - rm[j]);
        rs[j] += e;
        pbuf[(8 * hv + j) * 32 + 16 * t + col] = (_Float16)e;
      }
    }
    __builtin_amdgcn_wave_barrier();
    asm volatile("s_wait_dscnt 0" ::: "memory");

    AFrag ap;  // P as A fragment: rows = n, K = m (32)
    ap.h8[0] = *reinterpret_cast<const v8h*>(pbuf + col * 32 + 16 * hv);
    ap.h8[1] = *reinterpret_cast<const v8h*>(pbuf + col * 32 + 16 * hv + 8);

    // 16 accumulating WMMAs, B fragments from the LDS hh tile
    const _Float16* hb = &hh_lds[cur][0];
#pragma unroll
    for (int cb = 0; cb < 16; ++cb) {
      AFrag bh;
      const _Float16* lp = hb + (cb * 16 + col) * HH_STRIDE + 16 * hv;
      bh.h8[0] = *reinterpret_cast<const v8h*>(lp);
      bh.h8[1] = *reinterpret_cast<const v8h*>(lp + 8);
      oacc[cb] = wmma_f16(ap.v, bh.v, oacc[cb]);
    }

    wait_async();         // next tile finished loading (overlapped with above)
    __syncthreads();
    cur ^= 1;
  }

  // ---- normalize + epilogue ----
  float inv[8];
#pragma unroll
  for (int j = 0; j < 8; ++j) {
#pragma unroll
    for (int off = 1; off < 16; off <<= 1)
      rs[j] += __shfl_xor(rs[j], off, 16);
    inv[j] = 1.0f / rs[j];
  }
  const float gam = gamma[0];
#pragma unroll
  for (int cb = 0; cb < 16; ++cb) {
#pragma unroll
    for (int j = 0; j < 8; ++j) {
      int cc = cb * 16 + col;
      int nn = n0 + 8 * hv + j;
      size_t idx = ((size_t)b * C_ + cc) * N_ + nn;
      out[idx] = gam * oacc[cb][j] * inv[j] + x[idx];
    }
  }
}

// ---------------------------------------------------------------------------
extern "C" void kernel_launch(void* const* d_in, const int* in_sizes, int n_in,
                              void* d_out, int out_size, void* d_ws, size_t ws_size,
                              hipStream_t stream) {
  const float* x     = (const float*)d_in[0];
  const float* wf    = (const float*)d_in[1];
  const float* bf    = (const float*)d_in[2];
  const float* wg    = (const float*)d_in[3];
  const float* bg    = (const float*)d_in[4];
  const float* wh    = (const float*)d_in[5];
  const float* bh    = (const float*)d_in[6];
  const float* gamma = (const float*)d_in[7];
  float* out = (float*)d_out;

  // workspace layout (~19 MB)
  char* ws = (char*)d_ws;
  _Float16* xhT  = (_Float16*)ws;                         // B*N*C   f16 (8 MB)
  _Float16* W16  = xhT + (size_t)B_ * N_ * C_;            // 320*256 f16
  float*    b320 = (float*)(W16 + 320 * 256);             // 320     f32
  _Float16* fT   = (_Float16*)(b320 + 320);               // B*N*32  f16
  _Float16* gT   = fT + (size_t)B_ * N_ * C8_;            // B*N*32  f16
  _Float16* hh16 = gT + (size_t)B_ * N_ * C8_;            // B*C*N   f16 (8 MB)

  size_t total = (size_t)B_ * C_ * N_;
  cvt_x_kernel<<<dim3((unsigned)((total + 255) / 256)), dim3(256), 0, stream>>>(x, xhT);
  cvt_w_kernel<<<dim3((320 * 256 + 255) / 256), dim3(256), 0, stream>>>(
      wf, bf, wg, bg, wh, bh, W16, b320);

  dim3 grid(N_ / 64, B_);
  proj_kernel<<<grid, dim3(128), 0, stream>>>(xhT, W16, b320, fT, gT, hh16);
  attn_kernel<<<grid, dim3(128), 0, stream>>>(fT, gT, hh16, x, gamma, out);
}
